// TwoDimensionalSSMOptimized_26731876450906
// MI455X (gfx1250) — matrix operand order repaired and checked by reading, not compiled, not run on hardware
//
#include <hip/hip_runtime.h>

// ---------------- problem constants (from reference) ----------------
#define L32    32
#define SEQN   1024          // L32*L32
#define NDIM   16
#define KDIM   64            // kernel_dim
#define VPOW   64            // vander power dim
#define NR     16384         // R expansion terms
#define HCH    64            // channels per ssm
#define EMBED  1024
#define BSZ    8
#define BD     (BSZ * EMBED) // 8192

typedef __attribute__((ext_vector_type(16))) _Float16 v16h;
typedef __attribute__((ext_vector_type(8)))  float    v8f;

__device__ __forceinline__ v8f wmma_f16(v16h a, v16h b, v8f c) {
  // D = A(16x32 f16) * B(32x16 f16) + C(16x16 f32)
  return __builtin_amdgcn_wmma_f32_16x16x32_f16(
      /*neg_a=*/false, a, /*neg_b=*/false, b,
      /*c_mod=*/(short)0, c, /*reuse_a=*/false, /*reuse_b=*/false);
}

__device__ __forceinline__ float sigmoidf(float x) {
  return 1.0f / (1.0f + __expf(-x));
}

// ------------------------------------------------------------------
// K1: build vals = concat(sigmoid(A)^v, pad(sigmoid(B1),sigmoid(B2)))
// stored directly in WMMA B-fragment order (f16):
//   frag index = (((a*64 + ct)*2 + h)*32 + lane)*16 + t
//   lane: g=lane>>4, nl=lane&15 ; column c = ct*16+nl -> (k=c>>4, n=c&15)
//   element t covers row v = h*32 + g*16 + t     (contiguous k order, same as K2's A)
// ------------------------------------------------------------------
__global__ void k1_prep_vals(const float* __restrict__ A,
                             const float* __restrict__ B1,
                             const float* __restrict__ B2,
                             _Float16* __restrict__ vals_frag) {
  int idx = blockIdx.x * blockDim.x + threadIdx.x;   // 5*64*2*32*16 = 327680
  int t    = idx & 15;
  int lane = (idx >> 4) & 31;
  int hh   = (idx >> 9) & 1;
  int ct   = (idx >> 10) & 63;
  int a    = idx >> 16;
  if (a >= 5) return;
  int g  = lane >> 4;
  int nl = lane & 15;
  int c  = ct * 16 + nl;
  int k  = c >> 4;
  int n  = c & 15;
  int v  = hh * 32 + g * 16 + t;
  float val;
  if (a < 4) {
    float s = sigmoidf(A[(a * KDIM + k) * NDIM + n]);
    val = __powf(s, (float)v);                 // v==0 -> 1.0
  } else {
    if (v == 0)      val = sigmoidf(B1[k * NDIM + n]);
    else if (v == 1) val = sigmoidf(B2[k * NDIM + n]);
    else             val = 0.0f;
  }
  vals_frag[idx] = (_Float16)val;
}

// ------------------------------------------------------------------
// K2: prod[d,r,kn] = PROD_a GEMM_a(one_matrix[d,a] (R x V), vals[a] (V x KN))
// then reduce the 16 tile rows (= r2 expansion terms of one position p)
// One wave == one 16x16 output tile. 10 WMMAs per wave.
// Output w[d][p][kn] as f16.
// ------------------------------------------------------------------
__global__ void k2_coeff_gemm(const float* __restrict__ one_matrix,
                              const _Float16* __restrict__ vals_frag,
                              _Float16* __restrict__ w) {
  int lane = threadIdx.x & 31;
  int tile = blockIdx.x * (blockDim.x >> 5) + (threadIdx.x >> 5); // 131072
  int ct = tile & 63;
  int p  = (tile >> 6) & 1023;
  int d  = tile >> 16;

  int m = lane & 15;
  int g = lane >> 4;
  const v16h* bf = (const v16h*)vals_frag;

  v8f acc[5];
#pragma unroll
  for (int a = 0; a < 5; ++a) {
    const float* om =
        one_matrix + ((size_t)(d * 5 + a) * NR + (size_t)(p * 16 + m)) * VPOW;
    // prefetch the A-rows of the next p-tile (only HBM-cold stream here)
    if (p < 1023) __builtin_prefetch(om + 16 * VPOW, 0, 1);
    v16h alo, ahi;                         // contiguous k order: k = 16*g + t
#pragma unroll
    for (int t = 0; t < 16; ++t) {
      int K = t + 16 * g;
      alo[t] = (_Float16)om[K];
      ahi[t] = (_Float16)om[K + 32];
    }
    v16h blo = bf[((a * 64 + ct) * 2 + 0) * 32 + lane];
    v16h bhi = bf[((a * 64 + ct) * 2 + 1) * 32 + lane];
    v8f z = {};
    z = wmma_f16(alo, blo, z);
    z = wmma_f16(ahi, bhi, z);
    acc[a] = z;
  }
  v8f prod = acc[0] * acc[1] * acc[2] * acc[3] * acc[4];
  float sum = prod[0] + prod[1] + prod[2] + prod[3] +
              prod[4] + prod[5] + prod[6] + prod[7];      // M=0..7 (or 8..15)
  float tot = sum + __shfl_xor(sum, 16, 32);              // join half-waves
  if (lane < 16)
    w[((size_t)d * SEQN + p) * 1024 + ct * 16 + lane] = (_Float16)tot;
}

// ------------------------------------------------------------------
// K3: k[p][h][kd] = sum_{d,n} w[d][p][kd*16+n] * C_d[(h*64+kd)*16+n] * 0.25
// per kd: GEMM M=p(1024), N=h(64), K=(d,n)=32  -> 1 WMMA per tile
// ------------------------------------------------------------------
__global__ void k3_c_contract(const _Float16* __restrict__ w,
                              const float* __restrict__ C1,
                              const float* __restrict__ C2,
                              _Float16* __restrict__ k3) {
  int lane = threadIdx.x & 31;
  int wave = blockIdx.x * (blockDim.x >> 5) + (threadIdx.x >> 5); // 16384
  int ht = wave & 3;
  int pt = (wave >> 2) & 63;
  int kd = wave >> 8;

  int m = lane & 15;
  int g = lane >> 4;
  v16h av, bv;
#pragma unroll
  for (int t = 0; t < 16; ++t) {
    int K  = t + 16 * g;  // combined (d,n) for A
    int dd = K >> 4;
    int n  = K & 15;
    av[t] = w[((size_t)dd * SEQN + (pt * 16 + m)) * 1024 + kd * 16 + n];
    // B: K = g*16 + t  ->  d = g, n = t ; column h = ht*16 + (lane&15)
    const float* Cd = (g == 0) ? C1 : C2;
    bv[t] = (_Float16)(Cd[((size_t)(ht * 16 + m) * KDIM + kd) * NDIM + t] * 0.25f);
  }
  v8f z = {};
  z = wmma_f16(av, bv, z);
#pragma unroll
  for (int r = 0; r < 8; ++r) {             // C/D layout: M = r + 8g, N = lane&15
    int pp = pt * 16 + r + (g ? 8 : 0);
    int hh = ht * 16 + m;
    k3[((size_t)pp * HCH + hh) * KDIM + kd] = (_Float16)z[r];
  }
}

// ------------------------------------------------------------------
// K4: boundary scaling + 4-corner kernel flips summed into ksum[e][i*32+j]
// ------------------------------------------------------------------
__global__ void k4_ksum(const _Float16* __restrict__ k3,
                        _Float16* __restrict__ ksum) {
  int idx = blockIdx.x * blockDim.x + threadIdx.x;  // 1024*1024
  int e = idx >> 10;
  int s = idx & 1023;
  int i = s >> 5, j = s & 31;
  int h = e >> 4, ssm = e & 15;
  auto gat = [&](int ii, int jj, int dir) -> float {
    float sc = ((ii == 0) ? 2.0f : 1.0f) * ((jj == 0) ? 2.0f : 1.0f);
    if (ii == 0 && jj == 0) sc *= 0.25f;
    return (float)k3[((size_t)(ii * 32 + jj) * HCH + h) * KDIM + dir * 16 + ssm] * sc;
  };
  float v = gat(i, j, 0) + gat(31 - i, j, 1) +
            gat(i, 31 - j, 2) + gat(31 - i, 31 - j, 3);
  ksum[(size_t)e * 1024 + s] = (_Float16)v;
}

// ------------------------------------------------------------------
// K4x: convert x to f16 in (b,d,s) layout -> contiguous 2KB per channel
// image, so the conv kernel can stage it with async b128 LDS loads.
// ------------------------------------------------------------------
__global__ void k4x_xcast(const float* __restrict__ x,
                          _Float16* __restrict__ xh) {
  int idx = blockIdx.x * blockDim.x + threadIdx.x;  // 8388608
  int bd = idx & 8191;       // b*EMBED + d  (fastest -> coalesced reads)
  int s  = idx >> 13;
  xh[(size_t)bd * 1024 + s] = (_Float16)x[idx];
}

// ------------------------------------------------------------------
// K5: depthwise causal 2D conv + residual, one wave per (b,d).
// y[i][j] = sum_p ( X_shifted_by_p(32x32) * Toeplitz(k_row_p)(32x32) )
// LDS staging via GLOBAL_LOAD_ASYNC_TO_LDS_B128 (ASYNCcnt), then
// 32 p-steps x 4 WMMAs with fragments built from LDS.
// ------------------------------------------------------------------
__global__ void __launch_bounds__(32)
k5_conv(const _Float16* __restrict__ ksum,
        const _Float16* __restrict__ xh,
        const float* __restrict__ x,
        const float* __restrict__ omega,
        float* __restrict__ y) {
  __shared__ __align__(16) _Float16 sk[1024];
  __shared__ __align__(16) _Float16 sx[1024];
  int lane = threadIdx.x;
  int b = blockIdx.x >> 10;
  int d = blockIdx.x & 1023;

  // ---- async global -> LDS staging (2KB kernel tile + 2KB x tile) ----
  {
    unsigned skb = (unsigned)(unsigned long long)&sk[0];
    unsigned sxb = (unsigned)(unsigned long long)&sx[0];
    const char* gk = (const char*)(ksum + (size_t)d * 1024);
    const char* gx = (const char*)(xh + (size_t)(b * EMBED + d) * 1024);
#pragma unroll
    for (int it = 0; it < 4; ++it) {
      unsigned long long ga = (unsigned long long)(gk + it * 512 + lane * 16);
      unsigned la = skb + it * 512 + lane * 16;
      asm volatile("global_load_async_to_lds_b128 %0, %1, off"
                   :: "v"(la), "v"(ga) : "memory");
      unsigned long long gb = (unsigned long long)(gx + it * 512 + lane * 16);
      unsigned lb = sxb + it * 512 + lane * 16;
      asm volatile("global_load_async_to_lds_b128 %0, %1, off"
                   :: "v"(lb), "v"(gb) : "memory");
    }
    asm volatile("s_wait_asynccnt 0" ::: "memory");
  }
  __syncthreads();

  int ml = lane & 15;
  int g  = lane >> 4;
  v8f a00 = {}, a01 = {}, a10 = {}, a11 = {};
  for (int p = 0; p < 32; ++p) {
    v16h x0v, x1v, t0v, t1v;
#pragma unroll
    for (int t = 0; t < 16; ++t) {
      int K  = t + 16 * g;  // input column q'
      int r0 = ml - p;                          // shifted input row
      int r1 = 16 + ml - p;
      x0v[t] = (r0 >= 0) ? sx[r0 * 32 + K] : (_Float16)0.0f;
      x1v[t] = (r1 >= 0) ? sx[r1 * 32 + K] : (_Float16)0.0f;
      int q  = g * 16 + t;                      // Toeplitz row q'
      int j0 = ml, j1 = 16 + ml;
      t0v[t] = (j0 >= q) ? sk[p * 32 + (j0 - q)] : (_Float16)0.0f;
      t1v[t] = (j1 >= q) ? sk[p * 32 + (j1 - q)] : (_Float16)0.0f;
    }
    a00 = wmma_f16(x0v, t0v, a00);
    a01 = wmma_f16(x0v, t1v, a01);
    a10 = wmma_f16(x1v, t0v, a10);
    a11 = wmma_f16(x1v, t1v, a11);
  }
  float om = omega[d];
#pragma unroll
  for (int r = 0; r < 8; ++r) {
    int i0 = r + (g ? 8 : 0);
    int i1 = 16 + i0;
    int j0 = ml, j1 = 16 + ml;
    auto emit = [&](float acc, int ii, int jj) {
      size_t s = (size_t)(ii * 32 + jj);
      size_t off = s * BD + b * EMBED + d;
      y[off] = acc + x[off] * om;               // residual from f32 x
    };
    emit(a00[r], i0, j0);
    emit(a01[r], i0, j1);
    emit(a10[r], i1, j0);
    emit(a11[r], i1, j1);
  }
}

// ------------------------------------------------------------------
extern "C" void kernel_launch(void* const* d_in, const int* in_sizes, int n_in,
                              void* d_out, int out_size, void* d_ws, size_t ws_size,
                              hipStream_t stream) {
  (void)in_sizes; (void)n_in; (void)out_size; (void)ws_size;
  const float* x          = (const float*)d_in[0];
  const float* A          = (const float*)d_in[1];
  const float* B1         = (const float*)d_in[2];
  const float* B2         = (const float*)d_in[3];
  const float* C1         = (const float*)d_in[4];
  const float* C2         = (const float*)d_in[5];
  const float* omega      = (const float*)d_in[6];
  const float* one_matrix = (const float*)d_in[7];
  float* out = (float*)d_out;

  // workspace layout (bytes, 256-aligned):
  //   vals_frag f16 :        0 ..   655360
  //   w         f16 :   655360 ..  4849664   (2*1024*1024*2)
  //   k3        f16 :  4849664 .. 13238272   (1024*64*64*2)
  //   ksum      f16 : 13238272 .. 15335424   (1024*1024*2)
  //   xh        f16 : 15335424 .. 32112640   (8*1024*1024*2)
  char* ws = (char*)d_ws;
  _Float16* vals_frag = (_Float16*)(ws);
  _Float16* wbuf      = (_Float16*)(ws + 655360);
  _Float16* k3buf     = (_Float16*)(ws + 4849664);
  _Float16* ksum      = (_Float16*)(ws + 13238272);
  _Float16* xh        = (_Float16*)(ws + 15335424);

  hipLaunchKernelGGL(k1_prep_vals,  dim3(327680 / 256),  dim3(256), 0, stream,
                     A, B1, B2, vals_frag);
  hipLaunchKernelGGL(k2_coeff_gemm, dim3(131072 / 8),    dim3(256), 0, stream,
                     one_matrix, vals_frag, wbuf);
  hipLaunchKernelGGL(k3_c_contract, dim3(16384 / 8),     dim3(256), 0, stream,
                     wbuf, C1, C2, k3buf);
  hipLaunchKernelGGL(k4_ksum,       dim3(1048576 / 256), dim3(256), 0, stream,
                     k3buf, ksum);
  hipLaunchKernelGGL(k4x_xcast,     dim3(8388608 / 256), dim3(256), 0, stream,
                     x, xh);
  hipLaunchKernelGGL(k5_conv,       dim3(8192),          dim3(32),  0, stream,
                     ksum, xh, x, omega, out);
}
